// LSTM_16286515986443
// MI455X (gfx1250) — matrix operand order repaired and checked by reading, not compiled, not run on hardware
//
#include <hip/hip_runtime.h>
#include <hip/hip_bf16.h>
#include <stdint.h>
#include <stddef.h>

// LSTM: B=64, S=2048, D=256, H=256 on MI455X (gfx1250, wave32, WMMA).
#define B_  64
#define S_  2048
#define D_  256
#define H_  256
#define NG  1024   // 4 gates * H

#if defined(__has_builtin)
#if __has_builtin(__builtin_amdgcn_sched_barrier)
#define SCHED_FENCE() __builtin_amdgcn_sched_barrier(0)
#endif
#endif
#ifndef SCHED_FENCE
#define SCHED_FENCE()
#endif

typedef __attribute__((ext_vector_type(16))) __bf16 v16bf;
typedef __attribute__((ext_vector_type(8)))  __bf16 bf16x8;
typedef __attribute__((ext_vector_type(8)))  float  v8f;
typedef __attribute__((ext_vector_type(4)))  float  f32x4;
typedef __attribute__((ext_vector_type(4)))  unsigned short us4;
typedef __attribute__((ext_vector_type(8)))  unsigned short us8;

union Frag16 { v16bf v; bf16x8 h[2]; };

__device__ __forceinline__ unsigned short f2bf(float f) {
  union { float f; unsigned int u; } x; x.f = f;
  unsigned int u = x.u;
  u += 0x7fffu + ((u >> 16) & 1u);     // round-to-nearest-even
  return (unsigned short)(u >> 16);
}

__device__ __forceinline__ float sigmoidf_(float x) {
  return 1.0f / (1.0f + __expf(-x));
}

// A fragment: element groups at +0 and +16 ushorts.
__device__ __forceinline__ Frag16 ldfragA(const unsigned short* p) {
  Frag16 f;
  f.h[0] = *reinterpret_cast<const bf16x8*>(p);
  f.h[1] = *reinterpret_cast<const bf16x8*>(p + 16);
  return f;
}
// B fragment: element groups at +0 and +16 ushorts.
__device__ __forceinline__ Frag16 ldfragB(const unsigned short* p) {
  Frag16 f;
  f.h[0] = *reinterpret_cast<const bf16x8*>(p);
  f.h[1] = *reinterpret_cast<const bf16x8*>(p + 16);
  return f;
}

// ---------------------------------------------------------------------------
// Kernel 1: convert x (fp32) -> bf16, same layout [B*S, D]. Vectorized x4.
// ---------------------------------------------------------------------------
__global__ void k_cvt_x(const f32x4* __restrict__ x4,
                        us4* __restrict__ xb4, int n4) {
  int i = blockIdx.x * blockDim.x + threadIdx.x;
  int stride = gridDim.x * blockDim.x;
  for (; i < n4; i += stride) {
    f32x4 v = __builtin_nontemporal_load(&x4[i]);
    us4 o;
    o.x = f2bf(v.x); o.y = f2bf(v.y); o.z = f2bf(v.z); o.w = f2bf(v.w);
    xb4[i] = o;
  }
}

// ---------------------------------------------------------------------------
// Kernel 2: pack weights into WMMA B-operand layout Wpack[n][k] = W[k][h]
// (n = gate*256 + h), so each lane's B-fragment is two 16B runs along k. + biases.
// ---------------------------------------------------------------------------
__global__ void k_pack_w(const float* __restrict__ wii, const float* __restrict__ wif,
                         const float* __restrict__ wig, const float* __restrict__ wio,
                         const float* __restrict__ whi, const float* __restrict__ whf,
                         const float* __restrict__ whg, const float* __restrict__ who,
                         const float* __restrict__ bi,  const float* __restrict__ bfv,
                         const float* __restrict__ bg,  const float* __restrict__ bo,
                         unsigned short* __restrict__ w1p,
                         unsigned short* __restrict__ whp,
                         float* __restrict__ bcat) {
  int idx = blockIdx.x * blockDim.x + threadIdx.x;
  const int per = NG * D_;                        // 262144
  if (idx < per) {
    int n = idx / D_, k = idx % D_;
    int g = n >> 8, h = n & 255;
    const float* w = (g == 0) ? wii : (g == 1) ? wif : (g == 2) ? wig : wio;
    w1p[(size_t)n * D_ + k] = f2bf(w[(size_t)k * H_ + h]);
  } else if (idx < 2 * per) {
    int j = idx - per;
    int n = j / H_, k = j % H_;
    int g = n >> 8, h = n & 255;
    const float* w = (g == 0) ? whi : (g == 1) ? whf : (g == 2) ? whg : who;
    whp[(size_t)n * H_ + k] = f2bf(w[(size_t)k * H_ + h]);
  } else if (idx < 2 * per + NG) {
    int n = idx - 2 * per;
    int g = n >> 8, h = n & 255;
    const float* b = (g == 0) ? bi : (g == 1) ? bfv : (g == 2) ? bg : bo;
    bcat[n] = b[h];
  }
}

// ---------------------------------------------------------------------------
// Kernel 3: input-projection GEMM, bf16 WMMA, whole-kb double buffering
// enforced with sched_barrier fences: [prefetch kb+1] | fence | [8 WMMAs kb].
// Grid (1024, 8): block = 128 M-rows x 128 N-cols; all 8 waves share one
// 128-col W slab (64 KB = WGP$-resident). Per wave: 16x128 output tile.
// ---------------------------------------------------------------------------
__global__ void __launch_bounds__(256)
k_gemm_in(const unsigned short* __restrict__ xb,
          const unsigned short* __restrict__ w1p,
          const float* __restrict__ bias,
          float* __restrict__ gates) {
  const int lane  = threadIdx.x & 31;
  const int wave  = threadIdx.x >> 5;                // 0..7
  const int m0    = blockIdx.x * 128 + wave * 16;    // wave's M tile (m=b*S+s)
  const int n0    = blockIdx.y * 128;                // block's shared N slab

  const int hi      = (lane >= 16) ? 1 : 0;
  const int arow    = hi ? (lane - 16) : lane;       // A: local M row
  const int acolsel = hi ? 8 : 0;                    // A: K offset in 32-block
  const int bn      = lane & 15;                     // B/C: local N
  const int bksel   = hi ? 8 : 0;                    // B: K offset in 32-block

  const unsigned short* aptr = xb + (size_t)(m0 + arow) * D_ + acolsel;
  const unsigned short* bptr = w1p + (size_t)(n0 + bn) * D_ + bksel;

  v8f acc[8];
  #pragma unroll
  for (int t = 0; t < 8; ++t) acc[t] = v8f{};

  Frag16 a[2];
  Frag16 b[2][8];
  a[0] = ldfragA(aptr);
  #pragma unroll
  for (int t = 0; t < 8; ++t)
    b[0][t] = ldfragB(bptr + (size_t)t * 16 * D_);
  SCHED_FENCE();

  #pragma unroll
  for (int kb = 0; kb < 8; ++kb) {
    const int cur = kb & 1, nxt = cur ^ 1;
    if (kb < 7) {
      a[nxt] = ldfragA(aptr + (kb + 1) * 32);
      #pragma unroll
      for (int t = 0; t < 8; ++t)
        b[nxt][t] = ldfragB(bptr + (size_t)t * 16 * D_ + (kb + 1) * 32);
    }
    SCHED_FENCE();   // loads for kb+1 stay above; WMMAs for kb stay below
    #pragma unroll
    for (int t = 0; t < 8; ++t)
      acc[t] = __builtin_amdgcn_wmma_f32_16x16x32_bf16(
          false, a[cur].v, false, b[cur][t].v, (short)0, acc[t], false, false);
    SCHED_FENCE();
  }

  // Epilogue: +bias, scatter (non-temporal) to gates[s][b][n].
  #pragma unroll
  for (int t = 0; t < 8; ++t) {
    const int n = n0 + t * 16 + bn;
    const float bv = bias[n];
    #pragma unroll
    for (int j = 0; j < 8; ++j) {
      const int m = m0 + j + (hi ? 8 : 0);
      const int bb = m >> 11;                  // m / S
      const int s  = m & 2047;                 // m % S
      __builtin_nontemporal_store(acc[t][j] + bv,
                                  &gates[((size_t)s * B_ + bb) * NG + n]);
    }
  }
}

// ---------------------------------------------------------------------------
// Kernel 4: persistent recurrent scan. 4 blocks (16-batch slabs), 512 thr =
// 16 waves. Wave w owns h-columns [w*16,w*16+16) for ALL four gates ->
// register-only c/h update. Weights for gates i,f live in LDS (264 KB,
// bank-padded); gates g,o stream from L2 with fence-enforced whole-kb
// double buffering. Next-step gate inputs prefetch into the ping-pong
// accumulator buffer. One barrier per step.
// ---------------------------------------------------------------------------
#define SPAD 264   // padded row stride (ushorts): 528B rows -> 4-bank lane spread

__global__ void __launch_bounds__(512)
k_lstm_scan(const float* __restrict__ gates,
            const unsigned short* __restrict__ whp,
            float* __restrict__ out) {
  __shared__ unsigned short wlds[512 * SPAD];      // gates i,f weights: 264 KB
  __shared__ unsigned short hbuf[2][16 * SPAD];    // h double-buffer: 16.5 KB

  const int tid   = threadIdx.x;
  const int lane  = tid & 31;
  const int wave  = tid >> 5;            // 0..15
  const int hbase = wave * 16;           // h-column base within a gate
  const int bb    = blockIdx.x * 16;     // batch base

  const int hi      = (lane >= 16) ? 1 : 0;
  const int arow    = hi ? (lane - 16) : lane;
  const int acolsel = hi ? 8 : 0;
  const int bn      = lane & 15;
  const int bksel   = hi ? 8 : 0;

  // Zero h buffers.
  for (int i = tid; i < 2 * 16 * SPAD; i += 512)
    (&hbuf[0][0])[i] = 0;
  // Stage gates i,f weights into LDS (one padded row per thread).
  {
    const int n = tid;                                   // 0..511
    #pragma unroll
    for (int k = 0; k < H_; k += 8)
      *reinterpret_cast<us8*>(&wlds[n * SPAD + k]) =
          *reinterpret_cast<const us8*>(&whp[(size_t)n * H_ + k]);
  }

  float c[8], hreg[8];
  #pragma unroll
  for (int j = 0; j < 8; ++j) { c[j] = 0.0f; hreg[j] = 0.0f; }

  // B-fragment bases: gates 0,1 in LDS; gates 2,3 in global.
  const unsigned short* wl0 = &wlds[(0 * 256 + hbase + bn) * SPAD + bksel];
  const unsigned short* wl1 = &wlds[(1 * 256 + hbase + bn) * SPAD + bksel];
  const unsigned short* wg2 = whp + (size_t)(2 * 256 + hbase + bn) * H_ + bksel;
  const unsigned short* wg3 = whp + (size_t)(3 * 256 + hbase + bn) * H_ + bksel;

  // Ping-pong accumulator/input buffers; preload t=0 gate inputs.
  v8f xv[2][4];
  {
    const float* gp = gates + ((size_t)0 * B_ + bb) * NG;
    #pragma unroll
    for (int g = 0; g < 4; ++g)
      #pragma unroll
      for (int j = 0; j < 8; ++j) {
        const int m = j + (hi ? 8 : 0);
        xv[0][g][j] = __builtin_nontemporal_load(
            &gp[(size_t)m * NG + g * 256 + hbase + bn]);
      }
  }

  __syncthreads();

  for (int t = 0; t < S_; ++t) {
    const int p    = t & 1;
    const int bufR = t & 1;
    const int bufW = bufR ^ 1;

    // Global weight double-buffer (gates 2,3), whole-kb granularity.
    Frag16 g2[2], g3[2];
    g2[0] = ldfragB(wg2);
    g3[0] = ldfragB(wg3);
    SCHED_FENCE();

    // xv[p] += h(t) @ W_h  (A from LDS h, B from LDS (i,f) / L2 (g,o)).
    const unsigned short* hrow = &hbuf[bufR][arow * SPAD + acolsel];
    #pragma unroll
    for (int kb = 0; kb < 8; ++kb) {
      const int cur = kb & 1, nxt = cur ^ 1;
      if (kb < 7) {
        g2[nxt] = ldfragB(wg2 + (kb + 1) * 32);
        g3[nxt] = ldfragB(wg3 + (kb + 1) * 32);
      }
      SCHED_FENCE();   // kb+1 global loads above; kb WMMA group below
      const Frag16 a = ldfragA(hrow + kb * 32);
      {
        const Frag16 w = ldfragB(wl0 + kb * 32);
        xv[p][0] = __builtin_amdgcn_wmma_f32_16x16x32_bf16(
            false, a.v, false, w.v, (short)0, xv[p][0], false, false);
      }
      {
        const Frag16 w = ldfragB(wl1 + kb * 32);
        xv[p][1] = __builtin_amdgcn_wmma_f32_16x16x32_bf16(
            false, a.v, false, w.v, (short)0, xv[p][1], false, false);
      }
      xv[p][2] = __builtin_amdgcn_wmma_f32_16x16x32_bf16(
          false, a.v, false, g2[cur].v, (short)0, xv[p][2], false, false);
      xv[p][3] = __builtin_amdgcn_wmma_f32_16x16x32_bf16(
          false, a.v, false, g3[cur].v, (short)0, xv[p][3], false, false);
      SCHED_FENCE();
    }

    // Prefetch next step's gate inputs (issued after this step's global
    // weight loads so in-order LOADcnt waits for WMMAs don't block on them).
    {
      const int tn = (t + 1 < S_) ? (t + 1) : (S_ - 1);
      const float* gpn = gates + ((size_t)tn * B_ + bb) * NG;
      #pragma unroll
      for (int g = 0; g < 4; ++g)
        #pragma unroll
        for (int j = 0; j < 8; ++j) {
          const int m = j + (hi ? 8 : 0);
          xv[p ^ 1][g][j] = __builtin_nontemporal_load(
              &gpn[(size_t)m * NG + g * 256 + hbase + bn]);
        }
    }
    SCHED_FENCE();   // t+1 input loads stay ahead of the transcendental chain

    // Elementwise gate math, all in registers (same C-layout across gates).
    #pragma unroll
    for (int j = 0; j < 8; ++j) {
      const float iv = sigmoidf_(xv[p][0][j]);
      const float fv = sigmoidf_(xv[p][1][j]);
      const float gv = tanhf(xv[p][2][j]);
      const float ov = sigmoidf_(xv[p][3][j]);
      const float cv = fv * c[j] + iv * gv;
      c[j] = cv;
      const float hv = ov * tanhf(cv);
      hreg[j] = hv;
      const int m = j + (hi ? 8 : 0);
      hbuf[bufW][m * SPAD + hbase + bn] = f2bf(hv);
    }
    __syncthreads();
  }

  // Final state: out = [h_t (B*H) | c_t (B*H)], fp32.
  #pragma unroll
  for (int j = 0; j < 8; ++j) {
    const int m = j + (hi ? 8 : 0);
    const int b = bb + m;
    const int hcol = hbase + bn;
    out[(size_t)b * H_ + hcol] = hreg[j];
    out[(size_t)B_ * H_ + (size_t)b * H_ + hcol] = c[j];
  }
}

// ---------------------------------------------------------------------------
extern "C" void kernel_launch(void* const* d_in, const int* in_sizes, int n_in,
                              void* d_out, int out_size, void* d_ws, size_t ws_size,
                              hipStream_t stream) {
  (void)in_sizes; (void)n_in; (void)out_size; (void)ws_size;
  const float* x   = (const float*)d_in[0];
  const float* Wii = (const float*)d_in[1];
  const float* Whi = (const float*)d_in[2];
  const float* bi  = (const float*)d_in[3];
  const float* Wif = (const float*)d_in[4];
  const float* Whf = (const float*)d_in[5];
  const float* bfv = (const float*)d_in[6];
  const float* Wig = (const float*)d_in[7];
  const float* Whg = (const float*)d_in[8];
  const float* bg  = (const float*)d_in[9];
  const float* Wio = (const float*)d_in[10];
  const float* Who = (const float*)d_in[11];
  const float* bo  = (const float*)d_in[12];

  char* ws = (char*)d_ws;
  size_t ofs = 0;
  float* gates        = (float*)(ws + ofs);          ofs += (size_t)S_ * B_ * NG * 4;  // 512 MB
  unsigned short* xb  = (unsigned short*)(ws + ofs); ofs += (size_t)B_ * S_ * D_ * 2;  // 64 MB
  unsigned short* w1p = (unsigned short*)(ws + ofs); ofs += (size_t)NG * D_ * 2;       // 512 KB
  unsigned short* whp = (unsigned short*)(ws + ofs); ofs += (size_t)NG * H_ * 2;       // 512 KB
  float* bcat         = (float*)(ws + ofs);          ofs += (size_t)NG * 4;

  // 1) x -> bf16 (vectorized, non-temporal reads)
  k_cvt_x<<<4096, 256, 0, stream>>>((const f32x4*)x, (us4*)xb, (B_ * S_ * D_) / 4);

  // 2) pack weights (transposed, column-contiguous K) + biases
  {
    int total = 2 * NG * D_ + NG;
    k_pack_w<<<(total + 255) / 256, 256, 0, stream>>>(
        Wii, Wif, Wig, Wio, Whi, Whf, Whg, Who, bi, bfv, bg, bo, w1p, whp, bcat);
  }

  // 3) input-projection GEMM: [131072,256] @ [256,1024] -> gates (fp32)
  {
    dim3 grid((B_ * S_) / 128, NG / 128);
    k_gemm_in<<<grid, 256, 0, stream>>>(xb, w1p, bcat, gates);
  }

  // 4) batch-parallel persistent recurrence (4 WGPs, 1 barrier/step)
  k_lstm_scan<<<B_ / 16, 512, 0, stream>>>(gates, whp, (float*)d_out);
}
